// CustomRNN_64338610095028
// MI455X (gfx1250) — matrix operand order repaired and checked by reading, not compiled, not run on hardware
//
#include <hip/hip_runtime.h>
#include <hip/hip_bf16.h>

typedef __attribute__((ext_vector_type(16))) __bf16 v16bf;
typedef __attribute__((ext_vector_type(8)))  float  v8f;

#define B_SZ 256
#define T_SZ 2048
#define I_SZ 64
#define H_SZ 128
#define LDS_ROW 272   // bytes per h-row in LDS: 128*2 + 16 pad (keeps b128 loads aligned, spreads banks)

union ABReg  { v16bf v; __bf16 e[16]; uint4 q[2]; };
union AccReg { v8f v; float f[8]; };
union BFBits { __bf16 b; unsigned short u; };

// -------- Kernel 1: xu[row, h] = x[row, :] @ U + bias  (row = b*T + t flattened) --------
// grid: (B*T)/16 blocks of 256 threads; wave w handles N-tile [16w, 16w+16)
__global__ __launch_bounds__(256) void xu_gemm_kernel(
    const float* __restrict__ x, const float* __restrict__ U,
    const float* __restrict__ bias, float* __restrict__ xu)
{
    const int lane  = threadIdx.x & 31;
    const int wave  = threadIdx.x >> 5;
    const int grp   = lane >> 4;    // which 16-lane half
    const int nl    = lane & 15;
    const int hbase = wave * 16;
    const int m0    = blockIdx.x * 16;

    // ---- B operand (U) in WMMA B layout: lane half grp holds K in {8grp..8grp+7, 16+8grp..16+8grp+7} (matches A)
    ABReg Bt[2];
    #pragma unroll
    for (int kt = 0; kt < 2; ++kt) {
        #pragma unroll
        for (int e = 0; e < 16; ++e) {
            const int k = 32 * kt + 8 * grp + (e < 8 ? e : 8 + e);
            Bt[kt].e[e] = (__bf16)U[k * H_SZ + hbase + nl];
        }
    }

    // ---- A operand (x) in WMMA A layout: lane l & l+16 both row M=l;
    //      lane<16: K in {k0..k0+7, k0+16..k0+23} with k0 = 32*kt;  lane>=16: +8
    ABReg At[2];
    const float* xr = x + (long long)(m0 + nl) * I_SZ;
    #pragma unroll
    for (int kt = 0; kt < 2; ++kt) {
        const int k0 = 32 * kt + 8 * grp;
        float a0[8], a1[8];
        *(float4*)&a0[0] = *(const float4*)(xr + k0);
        *(float4*)&a0[4] = *(const float4*)(xr + k0 + 4);
        *(float4*)&a1[0] = *(const float4*)(xr + k0 + 16);
        *(float4*)&a1[4] = *(const float4*)(xr + k0 + 20);
        #pragma unroll
        for (int e = 0; e < 8; ++e) {
            At[kt].e[e]     = (__bf16)a0[e];
            At[kt].e[8 + e] = (__bf16)a1[e];
        }
    }

    // ---- C init = bias broadcast (C/D element N = lane&15 for all 8 VGPRs)
    AccReg acc;
    const float bv = bias[hbase + nl];
    #pragma unroll
    for (int r = 0; r < 8; ++r) acc.f[r] = bv;

    acc.v = __builtin_amdgcn_wmma_f32_16x16x32_bf16(false, At[0].v, false, Bt[0].v,
                                                    (short)0, acc.v, false, false);
    acc.v = __builtin_amdgcn_wmma_f32_16x16x32_bf16(false, At[1].v, false, Bt[1].v,
                                                    (short)0, acc.v, false, false);

    // ---- store D: element r -> row m0 + r + 8*grp, col hbase + nl
    #pragma unroll
    for (int r = 0; r < 8; ++r)
        xu[(long long)(m0 + r + 8 * grp) * H_SZ + hbase + nl] = acc.f[r];
}

// -------- Kernel 2: sequential scan, in-place over hid (which initially holds xu) --------
// grid: B/16 = 16 blocks of 256 threads (8 waves). WG owns batch rows [b0, b0+16);
// wave w owns hidden columns [16w, 16w+16). h ping-pongs through LDS in A-layout-friendly form.
__global__ __launch_bounds__(256) void rnn_scan_kernel(
    const float* __restrict__ W, float* __restrict__ hid, float* __restrict__ hlast)
{
    __shared__ __align__(16) unsigned char hbuf[2][16 * LDS_ROW];

    const int tid   = threadIdx.x;
    const int lane  = tid & 31;
    const int wave  = tid >> 5;
    const int grp   = lane >> 4;
    const int nl    = lane & 15;
    const int hbase = wave * 16;
    const int b0    = blockIdx.x * 16;

    // zero h_prev buffer for t=0
    for (int i = tid; i < (16 * LDS_ROW) / 4; i += 256)
        ((unsigned int*)hbuf[0])[i] = 0u;

    // ---- W in B layout, kept in registers for the whole scan (4 K-tiles of 32)
    ABReg Bt[4];
    #pragma unroll
    for (int kt = 0; kt < 4; ++kt) {
        #pragma unroll
        for (int e = 0; e < 16; ++e) {
            const int k = 32 * kt + 8 * grp + (e < 8 ? e : 8 + e);
            Bt[kt].e[e] = (__bf16)W[k * H_SZ + hbase + nl];
        }
    }

    // per-element global base index for this lane's 8 C/D slots (element r -> batch row r+8*grp)
    long long base[8];
    #pragma unroll
    for (int r = 0; r < 8; ++r)
        base[r] = (long long)(b0 + r + 8 * grp) * (long long)(T_SZ * H_SZ) + hbase + nl;

    // prefetch xu for t=0
    float cur[8];
    #pragma unroll
    for (int r = 0; r < 8; ++r) cur[r] = hid[base[r]];

    __syncthreads();

    for (int t = 0; t < T_SZ; ++t) {
        // prefetch next step's xu while WMMAs run
        float nxt[8];
        if (t + 1 < T_SZ) {
            #pragma unroll
            for (int r = 0; r < 8; ++r) nxt[r] = hid[base[r] + (long long)(t + 1) * H_SZ];
        }

        AccReg acc;
        #pragma unroll
        for (int r = 0; r < 8; ++r) acc.f[r] = cur[r];

        // A = h_prev from LDS: per lane two aligned 16B chunks per K-tile
        const unsigned char* hb = hbuf[t & 1] + nl * LDS_ROW;
        #pragma unroll
        for (int kt = 0; kt < 4; ++kt) {
            const int k0 = 32 * kt + 8 * grp;
            ABReg At;
            At.q[0] = *(const uint4*)(hb + k0 * 2);
            At.q[1] = *(const uint4*)(hb + (k0 + 16) * 2);
            acc.v = __builtin_amdgcn_wmma_f32_16x16x32_bf16(false, At.v, false, Bt[kt].v,
                                                            (short)0, acc.v, false, false);
        }

        // sigmoid, write h_t (overwrites xu slot), publish bf16 h to LDS for t+1
        unsigned char* ho = hbuf[(t + 1) & 1];
        #pragma unroll
        for (int r = 0; r < 8; ++r) {
            const float h = 1.0f / (1.0f + __expf(-acc.f[r]));
            hid[base[r] + (long long)t * H_SZ] = h;
            BFBits bb; bb.b = (__bf16)h;
            const int M = r + 8 * grp;
            *(unsigned short*)(ho + M * LDS_ROW + (hbase + nl) * 2) = bb.u;
            if (t == T_SZ - 1)
                hlast[(b0 + M) * H_SZ + hbase + nl] = h;
        }

        __syncthreads();

        #pragma unroll
        for (int r = 0; r < 8; ++r) cur[r] = nxt[r];
    }
}

extern "C" void kernel_launch(void* const* d_in, const int* in_sizes, int n_in,
                              void* d_out, int out_size, void* d_ws, size_t ws_size,
                              hipStream_t stream) {
    const float* x  = (const float*)d_in[0];
    const float* U  = (const float*)d_in[1];
    const float* W  = (const float*)d_in[2];
    const float* bf = (const float*)d_in[3];

    float* hlast = (float*)d_out;                         // [256, 128]
    float* hid   = (float*)d_out + (size_t)B_SZ * H_SZ;   // [256, 2048, 128], first holds xu

    dim3 blk(256);
    xu_gemm_kernel<<<dim3((B_SZ * T_SZ) / 16), blk, 0, stream>>>(x, U, bf, hid);
    rnn_scan_kernel<<<dim3(B_SZ / 16), blk, 0, stream>>>(W, hid, hlast);
}
